// classify_net_39307540693294
// MI455X (gfx1250) — compile-verified, matrix operand order checked by reading, not run on hardware
//
#include <hip/hip_runtime.h>
#include <hip/hip_bf16.h>

// ---------------------------------------------------------------------------
// GraphSAGE 2-layer inference for MI455X (gfx1250, wave32, WMMA).
// GEMMs use v_wmma_f32_16x16x32_bf16 (bf16 inputs, f32 accumulate) with a
// double-buffered, software-pipelined LDS staging loop (1 barrier per tile).
// ---------------------------------------------------------------------------

#define NNODES 50000
#define NEDGES 1600000
#define DIN    512
#define H1DIM  1550
#define H2DIM  512
#define NCLS   50
#define NGRAPH 64
#define EPSV   1e-5f

typedef __attribute__((ext_vector_type(16))) __bf16 v16bf;
typedef __attribute__((ext_vector_type(8)))  float  v8f;

// round-to-nearest-even f32 -> bf16 bits
static __device__ __forceinline__ unsigned short f32_bf16(float f) {
    union { float f; unsigned u; } v; v.f = f;
    unsigned r = v.u + 0x7FFFu + ((v.u >> 16) & 1u);
    return (unsigned short)(r >> 16);
}
static __device__ __forceinline__ unsigned packbf(float a, float b) {
    return (unsigned)f32_bf16(a) | ((unsigned)f32_bf16(b) << 16);
}

// ---------------------------------------------------------------------------
// Utility kernels
// ---------------------------------------------------------------------------
__global__ void zero_f32(float* __restrict__ p, size_t n) {
    size_t stride = (size_t)gridDim.x * blockDim.x;
    for (size_t i = (size_t)blockIdx.x * blockDim.x + threadIdx.x; i < n; i += stride)
        p[i] = 0.0f;
}

__global__ void deg_count(const long long* __restrict__ ei, float* __restrict__ deg, int E) {
    int e = blockIdx.x * blockDim.x + threadIdx.x;
    if (e < E) atomicAdd(&deg[(int)ei[E + e]], 1.0f);
}

// warp-per-edge scatter-add: sum[dst, :] += x[src, :]
__global__ void scatter_add_warp(const float* __restrict__ x,
                                 const long long* __restrict__ ei,
                                 float* __restrict__ sum, int D, int E) {
    int gw = (int)(((size_t)blockIdx.x * blockDim.x + threadIdx.x) >> 5);
    if (gw >= E) return;
    int lane = threadIdx.x & 31;
    size_t s = (size_t)ei[gw];
    size_t d = (size_t)ei[E + gw];
    const float* xs = x + s * (size_t)D;
    float* sd = sum + d * (size_t)D;
    if ((D & 3) == 0) {
        // rows are 16B-aligned only when D%4==0
        for (int f = lane * 4; f < D; f += 128) {
            float4 v = *(const float4*)(xs + f);
            atomicAdd(sd + f + 0, v.x);
            atomicAdd(sd + f + 1, v.y);
            atomicAdd(sd + f + 2, v.z);
            atomicAdd(sd + f + 3, v.w);
        }
    } else {
        for (int f = lane; f < D; f += 32) atomicAdd(sd + f, xs[f]);
    }
}

__global__ void scale_by_deg(float* __restrict__ a, const float* __restrict__ deg,
                             size_t total, int D) {
    size_t i = (size_t)blockIdx.x * blockDim.x + threadIdx.x;
    if (i >= total) return;
    float dg = deg[i / (size_t)D];
    a[i] = a[i] / fmaxf(dg, 1.0f);
}

// ---------------------------------------------------------------------------
// Fused SAGE GEMM:  out = relu(bn( Aagg@Wl + Aroot@Wr + bias ))
// Block: 256 threads (8 wave32), C tile 64(M) x 128(N), K staged 32-wide (bf16).
// Each wave computes a 32x32 C sub-tile = 2x2 grid of 16x16 WMMA accumulators.
// LDS A tile: row-major  [64 rows][40 pad] (k contiguous)  -> 2x ds_load_b128 / frag
// LDS B tile: col-major [128 cols][40 pad] (k contiguous)  -> 2x ds_load_b128 / frag
// 80B stride => bank advance 20 of 64: conflict-free within a 16-lane half.
// Interior blocks run a double-buffered pipeline: stage tile i+1 into the
// alternate LDS buffer while WMMAs consume tile i; one barrier per tile.
// ---------------------------------------------------------------------------
__global__ __launch_bounds__(256)
void sage_gemm_bn_relu(const float* __restrict__ Aagg, const float* __restrict__ Aroot,
                       const float* __restrict__ Wl,   const float* __restrict__ Wr,
                       const float* __restrict__ bias, const float* __restrict__ g,
                       const float* __restrict__ be,   const float* __restrict__ rm,
                       const float* __restrict__ rv,
                       float* __restrict__ out, int M, int K, int H) {
    constexpr int AST = 40;             // padded k-stride (elements) for A rows
    constexpr int BST = 40;             // padded k-stride (elements) for B cols
    constexpr int AELEMS = 64 * AST;    // one A buffer (2560 elems = 5120 B)
    constexpr int BELEMS = 128 * BST;   // one B buffer (5120 elems = 10240 B)
    __shared__ unsigned short ldsA[2 * AELEMS];
    __shared__ unsigned short ldsB[2 * BELEMS];

    const int tid  = threadIdx.x;
    const int lane = tid & 31;
    const int wave = tid >> 5;
    const int wm = wave & 1;   // 2 row groups of 32
    const int wn = wave >> 1;  // 4 col groups of 32
    const int mBase = blockIdx.y * 64;
    const int nBase = blockIdx.x * 128;

    const bool interior = (mBase + 64 <= M) && (nBase + 128 <= H);

    v8f acc[2][2] = {};

    const int kgrp = lane >> 4;     // 0 or 1 (lane half)
    const int lrow = lane & 15;

    // staging assignments (loop-invariant)
    const int ar  = tid >> 2;          // A: row 0..63
    const int ac0 = (tid & 3) * 8;     // A: k chunk {0,8,16,24}
    const int bcol = tid & 127;        // B: column 0..127
    const int bkh  = tid >> 7;         // B: k half 0..1 (16 k each)

    const bool aRowOK = (mBase + ar < M);
    const bool bColOK = (nBase + bcol < H);

    // loop-invariant LDS addresses (buffer 0; buffer 1 = +AELEMS/+BELEMS)
    unsigned short* const aDst = &ldsA[ar * AST + ac0];
    unsigned short* const bDst = &ldsB[bcol * BST + bkh * 16];
    const uint4* const apFrag[2] = {
        (const uint4*)&ldsA[(wm * 32 + 0  + lrow) * AST + kgrp * 8],
        (const uint4*)&ldsA[(wm * 32 + 16 + lrow) * AST + kgrp * 8]};
    const uint4* const bpFrag[2] = {
        (const uint4*)&ldsB[(wn * 32 + 0  + lrow) * BST + kgrp * 8],
        (const uint4*)&ldsB[(wn * 32 + 16 + lrow) * BST + kgrp * 8]};

    union Frag { v16bf v; uint4 q[2]; };

    // ---- one staged tile -> 4 WMMAs (buf is a literal 0/1 at call sites) ----
    auto compute_tile = [&](int buf) {
        const int aOff = buf * (AELEMS / 8);   // uint4 units
        const int bOff = buf * (BELEMS / 8);
        Frag fa[2], fb[2];
#pragma unroll
        for (int mi = 0; mi < 2; ++mi) {
            fa[mi].q[0] = apFrag[mi][aOff + 0];   // k = kgrp*8 + 0..7
            fa[mi].q[1] = apFrag[mi][aOff + 2];   // k = kgrp*8 + 16..23
        }
#pragma unroll
        for (int ni = 0; ni < 2; ++ni) {
            fb[ni].q[0] = bpFrag[ni][bOff + 0];
            fb[ni].q[1] = bpFrag[ni][bOff + 2];
        }
#pragma unroll
        for (int mi = 0; mi < 2; ++mi)
#pragma unroll
            for (int ni = 0; ni < 2; ++ni)
                acc[mi][ni] = __builtin_amdgcn_wmma_f32_16x16x32_bf16(
                    false, fa[mi].v, false, fb[ni].v,
                    (short)0, acc[mi][ni], false, false);
    };

    // ---- unguarded staging (row/col known valid, 32 k valid) ----
    // A rows may be only 8B-aligned (K=1550 -> 6200B stride), so use b64 loads.
    auto stageA_fast = [&](int buf, const float* src) {
        float2 f0 = *(const float2*)(src + 0);
        float2 f1 = *(const float2*)(src + 2);
        float2 f2 = *(const float2*)(src + 4);
        float2 f3 = *(const float2*)(src + 6);
        uint4 p;
        p.x = packbf(f0.x, f0.y);
        p.y = packbf(f1.x, f1.y);
        p.z = packbf(f2.x, f2.y);
        p.w = packbf(f3.x, f3.y);
        *(uint4*)(aDst + buf * AELEMS) = p;
        __builtin_prefetch(src + 64, 0, 1);   // speculative, 2 tiles ahead
    };
    auto stageB_fast = [&](int buf, const float* src) {  // stride-H column reads
        float v[16];
#pragma unroll
        for (int j = 0; j < 16; ++j) v[j] = src[(size_t)j * H];
        uint4 p0, p1;
        p0.x = packbf(v[0],  v[1]);  p0.y = packbf(v[2],  v[3]);
        p0.z = packbf(v[4],  v[5]);  p0.w = packbf(v[6],  v[7]);
        p1.x = packbf(v[8],  v[9]);  p1.y = packbf(v[10], v[11]);
        p1.z = packbf(v[12], v[13]); p1.w = packbf(v[14], v[15]);
        uint4* dst = (uint4*)(bDst + buf * BELEMS);
        dst[0] = p0;
        dst[1] = p1;
        __builtin_prefetch(src + (size_t)64 * H, 0, 1);
    };
    auto zeroA = [&](int buf) { *(uint4*)(aDst + buf * AELEMS) = uint4{0u, 0u, 0u, 0u}; };
    auto zeroB = [&](int buf) {
        uint4* dst = (uint4*)(bDst + buf * BELEMS);
        dst[0] = uint4{0u, 0u, 0u, 0u};
        dst[1] = uint4{0u, 0u, 0u, 0u};
    };

    const int Kfull = K & ~31;
    const int nt    = Kfull >> 5;   // # full 32-wide K tiles

    for (int phase = 0; phase < 2; ++phase) {
        const float* Asrc = phase ? Aroot : Aagg;
        const float* Wsrc = phase ? Wr : Wl;
        // per-thread base pointers, hoisted out of the K loop
        const float* aPtr = Asrc + (size_t)(mBase + ar) * K + ac0;
        const float* bPtr = Wsrc + (size_t)(bkh * 16) * H + (nBase + bcol);

        if (interior) {
            if (nt > 0) {
                // prologue: stage tile 0 into buffer 0
                stageA_fast(0, aPtr);
                stageB_fast(0, bPtr);
                __syncthreads();
                int i = 0;
                for (; i + 2 <= nt; i += 2) {
                    // stage tile i+1 into buf1 while computing tile i from buf0
                    stageA_fast(1, aPtr + (i + 1) * 32);
                    stageB_fast(1, bPtr + (size_t)((i + 1) * 32) * H);
                    compute_tile(0);
                    __syncthreads();
                    // stage tile i+2 into buf0 while computing tile i+1 from buf1
                    if (i + 2 < nt) {
                        stageA_fast(0, aPtr + (i + 2) * 32);
                        stageB_fast(0, bPtr + (size_t)((i + 2) * 32) * H);
                    }
                    compute_tile(1);
                    __syncthreads();
                }
                if (i < nt) {   // odd tile count: last tile already in buf0
                    compute_tile(0);
                    __syncthreads();
                }
            }
        } else {
            // edge blocks: simple single-buffer loop with row/col guards
            for (int k0 = 0; k0 < Kfull; k0 += 32) {
                if (aRowOK) stageA_fast(0, aPtr + k0); else zeroA(0);
                if (bColOK) stageB_fast(0, bPtr + (size_t)k0 * H); else zeroB(0);
                __syncthreads();
                compute_tile(0);
                __syncthreads();
            }
        }

        if (Kfull < K) {  // K tail tile (once per phase), per-element k guards
#pragma unroll
            for (int j = 0; j < 8; ++j) {
                int kk = Kfull + ac0 + j;
                float v = (aRowOK && kk < K) ? aPtr[Kfull + j] : 0.0f;
                aDst[j] = f32_bf16(v);
            }
#pragma unroll
            for (int j = 0; j < 16; ++j) {
                int kG = Kfull + bkh * 16 + j;
                float v = (bColOK && kG < K) ? bPtr[(size_t)(Kfull + j) * H] : 0.0f;
                bDst[j] = f32_bf16(v);
            }
            __syncthreads();
            compute_tile(0);
            __syncthreads();
        }
    }

    // ---- epilogue: bias + BN(eval) + ReLU, C/D lane layout ----
    // VGPR r: lanes 0-15 -> M=r, N=lane ; lanes 16-31 -> M=8+r, N=lane-16
    const int hiM = (lane >> 4) * 8;
#pragma unroll
    for (int ni = 0; ni < 2; ++ni) {
        int col = nBase + wn * 32 + ni * 16 + lrow;
        if (col >= H) continue;
        float scale = g[col] * rsqrtf(rv[col] + EPSV);
        float shift = be[col] - rm[col] * scale;
        float bcl   = bias[col];
#pragma unroll
        for (int mi = 0; mi < 2; ++mi) {
            int rowB = mBase + wm * 32 + mi * 16 + hiM;
#pragma unroll
            for (int r = 0; r < 8; ++r) {
                int row = rowB + r;
                if (row < M) {
                    float v = (acc[mi][ni][r] + bcl) * scale + shift;
                    out[(size_t)row * H + col] = fmaxf(v, 0.0f);
                }
            }
        }
    }
}

// ---------------------------------------------------------------------------
// Global mean pool (atomic) and FC + log_softmax
// ---------------------------------------------------------------------------
__global__ void pool_sum(const float* __restrict__ h, const long long* __restrict__ batch,
                         float* __restrict__ pooled, float* __restrict__ cnt, size_t total) {
    size_t i = (size_t)blockIdx.x * blockDim.x + threadIdx.x;
    if (i >= total) return;
    int n = (int)(i >> 9);        // D = 512
    int f = (int)(i & 511);
    int gph = (int)batch[n];
    atomicAdd(&pooled[gph * H2DIM + f], h[i]);
    if (f == 0) atomicAdd(&cnt[gph], 1.0f);
}

__global__ __launch_bounds__(128)
void fc_logsoftmax(const float* __restrict__ pooled, const float* __restrict__ cnt,
                   const float* __restrict__ fcw, const float* __restrict__ fcb,
                   float* __restrict__ out) {
    __shared__ float row[H2DIM];
    __shared__ float logits[NCLS];
    __shared__ float red[2];
    int gph = blockIdx.x;
    int tid = threadIdx.x;
    float inv = 1.0f / fmaxf(cnt[gph], 1.0f);
    for (int f = tid; f < H2DIM; f += blockDim.x)
        row[f] = pooled[gph * H2DIM + f] * inv;
    __syncthreads();
    if (tid < NCLS) {
        float acc = fcb[tid];
        for (int k = 0; k < H2DIM; ++k)
            acc += row[k] * fcw[k * NCLS + tid];
        logits[tid] = acc;
    }
    __syncthreads();
    if (tid == 0) {
        float mx = -1e30f;
        for (int c = 0; c < NCLS; ++c) mx = fmaxf(mx, logits[c]);
        float s = 0.0f;
        for (int c = 0; c < NCLS; ++c) s += __expf(logits[c] - mx);
        red[0] = mx;
        red[1] = logf(s);
    }
    __syncthreads();
    if (tid < NCLS) out[gph * NCLS + tid] = logits[tid] - red[0] - red[1];
}

// ---------------------------------------------------------------------------
// Host launcher
// ---------------------------------------------------------------------------
extern "C" void kernel_launch(void* const* d_in, const int* in_sizes, int n_in,
                              void* d_out, int out_size, void* d_ws, size_t ws_size,
                              hipStream_t stream) {
    const float*     x     = (const float*)d_in[0];
    const long long* ei    = (const long long*)d_in[1];
    const long long* batch = (const long long*)d_in[2];
    const float* W1l = (const float*)d_in[3];
    const float* W1r = (const float*)d_in[4];
    const float* b1  = (const float*)d_in[5];
    const float* g1  = (const float*)d_in[6];
    const float* be1 = (const float*)d_in[7];
    const float* rm1 = (const float*)d_in[8];
    const float* rv1 = (const float*)d_in[9];
    const float* W2l = (const float*)d_in[10];
    const float* W2r = (const float*)d_in[11];
    const float* b2  = (const float*)d_in[12];
    const float* g2  = (const float*)d_in[13];
    const float* be2 = (const float*)d_in[14];
    const float* rm2 = (const float*)d_in[15];
    const float* rv2 = (const float*)d_in[16];
    const float* fcw = (const float*)d_in[17];
    const float* fcb = (const float*)d_in[18];
    float* out = (float*)d_out;

    // ---- workspace layout (~723 MB) ----
    char*  ws  = (char*)d_ws;
    size_t off = 0;
    auto carve = [&](size_t bytes) -> float* {
        float* p = (float*)(ws + off);
        off += (bytes + 255) & ~(size_t)255;
        return p;
    };
    float* deg    = carve((size_t)NNODES * 4);
    float* cnt    = carve((size_t)NGRAPH * 4);
    float* pooled = carve((size_t)NGRAPH * H2DIM * 4);
    float* bufS   = carve((size_t)NNODES * DIN * 4);    // agg1, later h2
    float* h1     = carve((size_t)NNODES * H1DIM * 4);
    float* agg2   = carve((size_t)NNODES * H1DIM * 4);
    (void)ws_size;

    // ---- zero accumulation buffers ----
    zero_f32<<<1024, 256, 0, stream>>>(deg, (size_t)NNODES);
    zero_f32<<<64,   256, 0, stream>>>(cnt, (size_t)NGRAPH);
    zero_f32<<<256,  256, 0, stream>>>(pooled, (size_t)NGRAPH * H2DIM);
    zero_f32<<<4096, 256, 0, stream>>>(bufS, (size_t)NNODES * DIN);
    zero_f32<<<8192, 256, 0, stream>>>(agg2, (size_t)NNODES * H1DIM);

    // ---- degree ----
    deg_count<<<(NEDGES + 255) / 256, 256, 0, stream>>>(ei, deg, NEDGES);

    // ---- layer 1: mean aggregation of x ----
    scatter_add_warp<<<(NEDGES + 7) / 8, 256, 0, stream>>>(x, ei, bufS, DIN, NEDGES);
    {
        size_t total = (size_t)NNODES * DIN;
        scale_by_deg<<<(unsigned)((total + 255) / 256), 256, 0, stream>>>(bufS, deg, total, DIN);
    }
    // ---- layer 1 GEMM (WMMA bf16): h1 = relu(bn(agg@W1l + x@W1r + b1)) ----
    {
        dim3 grid((H1DIM + 127) / 128, (NNODES + 63) / 64);
        sage_gemm_bn_relu<<<grid, 256, 0, stream>>>(bufS, x, W1l, W1r,
                                                    b1, g1, be1, rm1, rv1,
                                                    h1, NNODES, DIN, H1DIM);
    }

    // ---- layer 2: mean aggregation of h1 ----
    scatter_add_warp<<<(NEDGES + 7) / 8, 256, 0, stream>>>(h1, ei, agg2, H1DIM, NEDGES);
    {
        size_t total = (size_t)NNODES * H1DIM;
        scale_by_deg<<<(unsigned)((total + 255) / 256), 256, 0, stream>>>(agg2, deg, total, H1DIM);
    }
    // ---- layer 2 GEMM: h2 (reuses bufS) ----
    {
        dim3 grid((H2DIM + 127) / 128, (NNODES + 63) / 64);
        sage_gemm_bn_relu<<<grid, 256, 0, stream>>>(agg2, h1, W2l, W2r,
                                                    b2, g2, be2, rm2, rv2,
                                                    bufS, NNODES, H1DIM, H2DIM);
    }

    // ---- global mean pool + classifier ----
    {
        size_t total = (size_t)NNODES * H2DIM;
        pool_sum<<<(unsigned)((total + 255) / 256), 256, 0, stream>>>(bufS, batch, pooled, cnt, total);
    }
    fc_logsoftmax<<<NGRAPH, 128, 0, stream>>>(pooled, cnt, fcw, fcb, out);
}
